// PcheckNormalizedInnerProductWithAutoScaleT1_67765993997127
// MI455X (gfx1250) — compile-verified, matrix operand order checked by reading, not run on hardware
//
#include <hip/hip_runtime.h>
#include <hip/hip_bf16.h>
#include <math.h>

// Problem sizes (fixed by the reference)
#define B_    2048
#define D_    512
#define C_    10572
#define C_PAD 10624   // 64 * 166, zero-padded weight rows for clean WMMA tiling

typedef __attribute__((ext_vector_type(16))) __bf16 v16bf;
typedef __attribute__((ext_vector_type(8)))  float  v8f;

union Frag16 {
    uint4 u[2];   // two 16B chunks: K {0..7 | 8..15} and K {16..23 | 24..31}
    v16bf v;
};

// ---------------------------------------------------------------------------
// Phase 1: L2-normalize rows of a [rows, D] fp32 matrix into bf16 (row-major).
// Rows >= nValid are written as zeros (padding for the weight matrix).
// ---------------------------------------------------------------------------
__global__ __launch_bounds__(256)
void row_normalize_bf16(const float* __restrict__ src, unsigned short* __restrict__ dst,
                        int nValid) {
    const int row = blockIdx.x;
    __shared__ float red[256];

    float ss = 0.0f;
    const float* p = src + (size_t)row * D_;
    if (row < nValid) {
        for (int j = threadIdx.x; j < D_; j += 256) {
            float v = p[j];
            ss += v * v;
        }
    }
    red[threadIdx.x] = ss;
    __syncthreads();
    for (int s = 128; s > 0; s >>= 1) {
        if (threadIdx.x < s) red[threadIdx.x] += red[threadIdx.x + s];
        __syncthreads();
    }
    const float rn = rsqrtf(red[0]);

    unsigned short* q = dst + (size_t)row * D_;
    if (row < nValid) {
        for (int j = threadIdx.x; j < D_; j += 256) {
            __hip_bfloat16 h = __float2bfloat16(p[j] * rn);
            q[j] = *reinterpret_cast<unsigned short*>(&h);
        }
    } else {
        for (int j = threadIdx.x; j < D_; j += 256) q[j] = 0;
    }
}

// ---------------------------------------------------------------------------
// Phase 2: cos[B, C] = nf[B, D] (bf16) @ nw[C_PAD, D]^T (bf16), f32 accumulate
// via v_wmma_f32_16x16x32_bf16.
//
// Block = 256 threads = 8 waves covering a 128(M) x 64(N) tile.
//  - The 64 nw rows (contiguous 64 KB) are staged once into LDS and shared by
//    all 8 waves (B fragments come from ds_load_b128, not redundant global).
//  - Each wave owns a 16(M) x 64(N) strip; its A fragment is software-
//    pipelined one k-step ahead so global-load latency hides under the
//    4 WMMAs + LDS reads of the current step.
// ---------------------------------------------------------------------------
__global__ __launch_bounds__(256)
void cosine_gemm_wmma(const unsigned short* __restrict__ nf,
                      const unsigned short* __restrict__ nw,
                      float* __restrict__ cosOut) {
    __shared__ __align__(16) unsigned short ldsB[64 * D_];   // 64 KB

    const int wave = threadIdx.x >> 5;
    const int lane = threadIdx.x & 31;

    const int m0 = blockIdx.y * 128 + wave * 16;   // 16 rows of nf
    const int n0 = blockIdx.x * 64;                // 64 rows of nw (columns of cos)

    // Cooperative contiguous copy: nw rows n0..n0+63 -> LDS (4096 x 16B)
    {
        const uint4* src = (const uint4*)(nw + (size_t)n0 * D_);
        uint4*       dst = (uint4*)ldsB;
        #pragma unroll
        for (int i = 0; i < 16; ++i) {
            dst[threadIdx.x + 256 * i] = src[threadIdx.x + 256 * i];
        }
    }
    __syncthreads();

    const int lm    = lane & 15;          // row within the 16-row group
    const int khalf = (lane >> 4) << 3;   // lanes 0-15 -> K base 0, lanes 16-31 -> K base 8

    const unsigned short* aRow = nf + (size_t)(m0 + lm) * D_ + khalf;
    const unsigned short* lb0 = &ldsB[(size_t)( 0 + lm) * D_ + khalf];
    const unsigned short* lb1 = &ldsB[(size_t)(16 + lm) * D_ + khalf];
    const unsigned short* lb2 = &ldsB[(size_t)(32 + lm) * D_ + khalf];
    const unsigned short* lb3 = &ldsB[(size_t)(48 + lm) * D_ + khalf];

    v8f acc0 = {}, acc1 = {}, acc2 = {}, acc3 = {};

    // Prologue: A fragment for k = 0
    Frag16 aCur, aNext;
    aCur.u[0] = *(const uint4*)(aRow);
    aCur.u[1] = *(const uint4*)(aRow + 16);

    #pragma unroll
    for (int k = 0; k < D_; k += 32) {
        // Prefetch next A fragment (global) under the compute of this step
        if (k + 32 < D_) {
            aNext.u[0] = *(const uint4*)(aRow + k + 32);
            aNext.u[1] = *(const uint4*)(aRow + k + 48);
        }

        // B fragments from LDS (shared across the 8 waves of the block)
        Frag16 b0, b1, b2, b3;
        b0.u[0] = *(const uint4*)(lb0 + k);
        b0.u[1] = *(const uint4*)(lb0 + k + 16);
        b1.u[0] = *(const uint4*)(lb1 + k);
        b1.u[1] = *(const uint4*)(lb1 + k + 16);
        b2.u[0] = *(const uint4*)(lb2 + k);
        b2.u[1] = *(const uint4*)(lb2 + k + 16);
        b3.u[0] = *(const uint4*)(lb3 + k);
        b3.u[1] = *(const uint4*)(lb3 + k + 16);

        acc0 = __builtin_amdgcn_wmma_f32_16x16x32_bf16(false, aCur.v, false, b0.v,
                                                       (short)0, acc0, false, false);
        acc1 = __builtin_amdgcn_wmma_f32_16x16x32_bf16(false, aCur.v, false, b1.v,
                                                       (short)0, acc1, false, false);
        acc2 = __builtin_amdgcn_wmma_f32_16x16x32_bf16(false, aCur.v, false, b2.v,
                                                       (short)0, acc2, false, false);
        acc3 = __builtin_amdgcn_wmma_f32_16x16x32_bf16(false, aCur.v, false, b3.v,
                                                       (short)0, acc3, false, false);

        aCur = aNext;
    }

    // C/D layout: lane -> column, VGPR i -> row (i + 8 for lanes 16..31)
    const int rofs = (lane >> 4) << 3;
    const v8f accs[4] = {acc0, acc1, acc2, acc3};
    #pragma unroll
    for (int t = 0; t < 4; ++t) {
        const int col = n0 + t * 16 + lm;
        if (col < C_) {
            #pragma unroll
            for (int i = 0; i < 8; ++i) {
                cosOut[(size_t)(m0 + rofs + i) * C_ + col] = accs[t][i];
            }
        }
    }
}

// ---------------------------------------------------------------------------
// Phase 3: per-row softmax stats, margin math, and new_logits write.
// One block per row. Row (42 KB) stays hot in L2 between the two passes.
// new_logits is write-only -> nontemporal stores (don't evict cos from L2).
// ---------------------------------------------------------------------------
__global__ __launch_bounds__(256)
void row_softmax_fix(const float* __restrict__ cosM, const int* __restrict__ label,
                     float* __restrict__ newLogits, float* __restrict__ pgt,
                     float* __restrict__ bval, float scale) {
    const int row = blockIdx.x;
    const float* crow = cosM + (size_t)row * C_;
    __shared__ float red[256];
    __shared__ float bcast;

    // pass 1: row max of logits
    float mx = -3.402823466e+38f;
    for (int j = threadIdx.x; j < C_; j += 256) mx = fmaxf(mx, crow[j]);
    red[threadIdx.x] = mx;
    __syncthreads();
    for (int s = 128; s > 0; s >>= 1) {
        if (threadIdx.x < s) red[threadIdx.x] = fmaxf(red[threadIdx.x], red[threadIdx.x + s]);
        __syncthreads();
    }
    const float m = red[0] * scale;   // max logit
    __syncthreads();

    // pass 2: shifted sum of exp
    float se = 0.0f;
    for (int j = threadIdx.x; j < C_; j += 256) se += __expf(scale * crow[j] - m);
    red[threadIdx.x] = se;
    __syncthreads();
    for (int s = 128; s > 0; s >>= 1) {
        if (threadIdx.x < s) red[threadIdx.x] += red[threadIdx.x + s];
        __syncthreads();
    }
    const float sumexp = red[0];

    const int lab = label[row];
    if (threadIdx.x == 0) {
        const float cg = crow[lab];
        const float eg = __expf(scale * cg - m);
        const float p  = eg / sumexp;
        // B = sum(exp(logits)) - exp(logit_gt) == exp(m) * (sumexp_shifted - eg)
        const float Bv = (sumexp - eg) * __expf(m);
        const float theta = acosf(cg);
        const float PIO2 = 1.57079632679489662f;
        const float dt = (theta >= PIO2) ? (PIO2 - 0.01f) : theta;
        const float s = (1.0f / cg) * logf(Bv * (3.14159265358979323f / (2.0f * dt) - 1.0f));
        pgt[row]  = p;
        bval[row] = Bv;
        bcast = s * cg;
    }
    __syncthreads();
    const float scos = bcast;

    float* nrow = newLogits + (size_t)row * C_;
    for (int j = threadIdx.x; j < C_; j += 256) {
        const float v = (j == lab) ? scos : scale * crow[j];
        __builtin_nontemporal_store(v, nrow + j);
    }
}

// ---------------------------------------------------------------------------
// Phase 4: reduce per-row stats to the 6 output scalars.
// ---------------------------------------------------------------------------
__global__ __launch_bounds__(256)
void finalize_stats(const float* __restrict__ pgt, const float* __restrict__ bval,
                    float* __restrict__ scalars, float scale) {
    __shared__ float s_sum[256], s_min[256], s_max[256], s_bsum[256];
    float su = 0.0f, mn = 3.402823466e+38f, mxv = -3.402823466e+38f, bs = 0.0f;
    for (int i = threadIdx.x; i < B_; i += 256) {
        const float p = pgt[i];
        su += p;
        mn = fminf(mn, p);
        mxv = fmaxf(mxv, p);
        bs += bval[i];
    }
    s_sum[threadIdx.x] = su;  s_min[threadIdx.x] = mn;
    s_max[threadIdx.x] = mxv; s_bsum[threadIdx.x] = bs;
    __syncthreads();
    for (int s = 128; s > 0; s >>= 1) {
        if (threadIdx.x < s) {
            s_sum[threadIdx.x]  += s_sum[threadIdx.x + s];
            s_min[threadIdx.x]   = fminf(s_min[threadIdx.x], s_min[threadIdx.x + s]);
            s_max[threadIdx.x]   = fmaxf(s_max[threadIdx.x], s_max[threadIdx.x + s]);
            s_bsum[threadIdx.x] += s_bsum[threadIdx.x + s];
        }
        __syncthreads();
    }
    if (threadIdx.x == 0) {
        const float avg_p = s_sum[0] / (float)B_;
        const float b_avg = s_bsum[0] / (float)B_;
        const float PI_F = 3.14159265358979323f;
        const float B_avg = acosf(logf(b_avg / (float)(C_ - 1)) / scale) / PI_F * 180.0f;
        scalars[0] = avg_p;
        scalars[1] = s_min[0];
        scalars[2] = s_max[0];
        scalars[3] = B_avg;
        scalars[4] = 0.0f;
        scalars[5] = 0.0f;
    }
}

// ---------------------------------------------------------------------------
// Launcher
// ---------------------------------------------------------------------------
extern "C" void kernel_launch(void* const* d_in, const int* in_sizes, int n_in,
                              void* d_out, int out_size, void* d_ws, size_t ws_size,
                              hipStream_t stream) {
    (void)in_sizes; (void)n_in; (void)out_size; (void)ws_size;

    const float* feat    = (const float*)d_in[0];   // [B, D] fp32
    const int*   label   = (const int*)  d_in[1];   // [B]
    const float* weights = (const float*)d_in[2];   // [C, D] fp32

    float* out       = (float*)d_out;
    float* cosOut    = out;                                // [B, C]
    float* newLogits = out + (size_t)B_ * C_;              // [B, C]
    float* scalars   = out + 2 * (size_t)B_ * C_;          // [6]

    // Workspace layout
    unsigned short* nf = (unsigned short*)d_ws;            // [B, D] bf16
    unsigned short* nw = nf + (size_t)B_ * D_;             // [C_PAD, D] bf16
    float* pgt  = (float*)(nw + (size_t)C_PAD * D_);       // [B]
    float* bval = pgt + B_;                                // [B]

    const float scale = sqrtf(2.0f) * logf((float)(C_ - 1));

    // Phase 1: normalize + cast to bf16 (weights padded with zero rows)
    row_normalize_bf16<<<B_,    256, 0, stream>>>(feat,    nf, B_);
    row_normalize_bf16<<<C_PAD, 256, 0, stream>>>(weights, nw, C_);

    // Phase 2: WMMA GEMM -> cos
    dim3 grid(C_PAD / 64, B_ / 128);
    cosine_gemm_wmma<<<grid, 256, 0, stream>>>(nf, nw, cosOut);

    // Phase 3: softmax stats + new_logits
    row_softmax_fix<<<B_, 256, 0, stream>>>(cosOut, label, newLogits, pgt, bval, scale);

    // Phase 4: scalar outputs
    finalize_stats<<<1, 256, 0, stream>>>(pgt, bval, scalars, scale);
}
